// ESN_2D_82764019794627
// MI455X (gfx1250) — matrix operand order repaired and checked by reading, not compiled, not run on hardware
//
#include <hip/hip_runtime.h>
#include <hip/hip_bf16.h>

// ===========================================================================
// 2D ESN on MI455X (gfx1250):
//  - recurrence as wavefront over 127 anti-diagonals; per-pixel (64x512)@(512x256)
//    GEMM via v_wmma_f32_16x16x32_f16; weights pre-packed in B-fragment layout,
//    staged into LDS (256 KB of the 320 KB/WGP budget) with
//    global_load_async_to_lds_b128 + s_wait_asynccnt (ASYNCcnt path)
//  - HtH Gram matrix per batch via WMMA; Haug columns resolved on the fly from
//    the f16 state buffer; each wave amortizes one A-fragment over 4 tiles
//  - ridge solve: f32 Gaussian elimination, one workgroup per batch
// Workspace use (~206 MB): state f16 (128 MB) | packed W1||W2 | HtH f32 | HtU f32
// ===========================================================================

typedef __attribute__((ext_vector_type(16))) _Float16 v16h;
typedef __attribute__((ext_vector_type(8)))  float    v8f;

#define B_    64
#define HH    64
#define WW    64
#define NR    256
#define NPIX  (HH*WW)
#define LVAL  3969        // (H-1)*(W-1)
#define DDIM  513         // 2n+1
#define DPAD  528         // 33 tiles of 16
#define KB_TOT 125        // ceil(3969/32)
#define ALPHA2 25.0f

union PkU   { unsigned u;    _Float16 h[2]; };
union FragU { unsigned u[8]; v16h v; };

static __device__ inline v8f vzero8() {
    v8f z;
#pragma unroll
    for (int e = 0; e < 8; ++e) z[e] = 0.f;
    return z;
}

// A-fragment K index for (v, lane-half): VGPR0-7 hold K 0..15 / 16..31
// (contiguous order, matching the packed B fragments).
static __device__ inline int kmapA(int v, int half) {
    return half * 16 + 2 * v;
}

// Flat->LDS mapping truncates to addr[31:0] (ISA 10.2), so the low 32 bits of
// a generic pointer to __shared__ are the wave-relative LDS byte address.
static __device__ inline unsigned lds_addr32(const void* p) {
    return (unsigned)(size_t)p;
}

// ---------------------------------------------------------------------------
// Pack W1/W2 (f32 row-major 256x256, W[k][m]) into the WMMA B-fragment lane
// layout: pk[((kb*16+nt)*32+lane)*8+v] = {W[kb*32+h*16+2v][nt*16+n],
//                                         W[.. +1][..]} with n=lane%16,h=lane/16
// ---------------------------------------------------------------------------
__global__ __launch_bounds__(256) void pack_weights(
    const float* __restrict__ W1, const float* __restrict__ W2,
    unsigned* __restrict__ W1pk, unsigned* __restrict__ W2pk)
{
    int idx  = blockIdx.x * blockDim.x + threadIdx.x;   // 0..65535
    int w    = idx >> 15;
    int r    = idx & 32767;
    int kb   = r >> 12;
    int nt   = (r >> 8) & 15;
    int lane = (r >> 3) & 31;
    int v    = r & 7;
    int n    = nt * 16 + (lane & 15);
    int k    = kb * 32 + (lane >> 4) * 16 + 2 * v;
    const float* src = w ? W2 : W1;
    PkU p;
    p.h[0] = (_Float16)src[(size_t)k * NR + n];
    p.h[1] = (_Float16)src[(size_t)(k + 1) * NR + n];
    (w ? W2pk : W1pk)[r] = p.u;
}

// ---------------------------------------------------------------------------
// One anti-diagonal step. One workgroup (8 waves) per pixel; wave (bt,g)
// owns batch-tile bt and n-tiles g*8..g*8+7. K = 256 (left@W1) + 256 (up@W2).
// W1pk must be immediately followed by W2pk in memory (contiguous 256 KB).
// ---------------------------------------------------------------------------
__global__ __launch_bounds__(256) void esn_diag(
    const float* __restrict__ x, const float* __restrict__ Win,
    const unsigned* __restrict__ W1pk, const unsigned* __restrict__ W2pk,
    _Float16* __restrict__ S, int diag, int i0)
{
    __shared__ __align__(32) unsigned ldsW[2 * 8 * 16 * 32 * 8];  // 256 KB
    const int tid = threadIdx.x;

    // Async-copy the contiguous 256 KB packed W1||W2 straight into LDS:
    // 256 threads x 64 iters x 16 B/lane, tracked by ASYNCcnt (no VGPR bounce).
    {
        const unsigned ldsBase = lds_addr32(ldsW);
        (void)W2pk; // contiguous with W1pk; copied as one region
        for (int it = 0; it < 64; ++it) {
            const unsigned off = (unsigned)(tid * 16) + (unsigned)(it * 4096);
            const unsigned la  = ldsBase + off;
            asm volatile("global_load_async_to_lds_b128 %0, %1, %2"
                         :: "v"(la), "v"(off), "s"(W1pk) : "memory");
        }
        asm volatile("s_wait_asynccnt 0" ::: "memory");
    }
    __syncthreads();

    const int i = i0 + blockIdx.x;
    const int j = diag - i;
    const int pix = i * WW + j;

    const int wave  = tid >> 5, lane = tid & 31;
    const int bt    = wave & 3;        // batch tile 0..3
    const int g     = wave >> 2;       // n-tile group 0..1
    const int laneN = lane & 15;
    const int half  = lane >> 4;
    const int mofs  = half * 8;
    const int bRow  = bt * 16 + laneN; // A-matrix row = batch index

    v8f acc[8];
#pragma unroll
    for (int t = 0; t < 8; ++t) acc[t] = vzero8();

    const _Float16* leftBase = (j > 0)
        ? S + (size_t)((i * WW + (j - 1)) * B_) * NR : nullptr;
    const _Float16* upBase = (i > 0)
        ? S + (size_t)(((i - 1) * WW + j) * B_) * NR : nullptr;

    for (int pass = 0; pass < 2; ++pass) {
        const _Float16* sb = pass ? upBase : leftBase;
        if (!sb) continue;
        const unsigned* lw   = ldsW + pass * 32768;
        const unsigned* rowp = (const unsigned*)(sb + (size_t)bRow * NR);
        for (int kb = 0; kb < 8; ++kb) {
            FragU af;
#pragma unroll
            for (int v = 0; v < 8; ++v)
                af.u[v] = rowp[(kb * 32 + kmapA(v, half)) >> 1];
            if (kb < 7)
                __builtin_prefetch((const void*)(rowp + (((kb + 1) * 32) >> 1)), 0, 1);
#pragma unroll
            for (int t = 0; t < 8; ++t) {
                const int nt = g * 8 + t;
                const v16h bf = *(const v16h*)(lw + ((size_t)(kb * 16 + nt) * 32 + lane) * 8);
                acc[t] = __builtin_amdgcn_wmma_f32_16x16x32_f16(
                    false, af.v, false, bf, (short)0, acc[t], false, false);
            }
        }
    }

    // bias (x * W_in), tanh, store f16 state
    float xv[8];
#pragma unroll
    for (int vr = 0; vr < 8; ++vr) {
        const int b = bt * 16 + mofs + vr;
        xv[vr] = x[(size_t)(b * HH + i) * WW + j];
    }
    _Float16* outp = S + (size_t)(pix * B_) * NR;
#pragma unroll
    for (int t = 0; t < 8; ++t) {
        const int n = (g * 8 + t) * 16 + laneN;
        const float wn = Win[n];
#pragma unroll
        for (int vr = 0; vr < 8; ++vr) {
            const float pre = acc[t][vr] + xv[vr] * wn;
            const int b = bt * 16 + mofs + vr;
            outp[(size_t)b * NR + n] = (_Float16)tanhf(pre);
        }
    }
}

// ---------------------------------------------------------------------------
// Haug(b, l, c): row l -> pixel (i=1+l/63, j=1+l%63); cols 0..255 = left state,
// 256..511 = up state, 512 = ones, >512 / l>=LVAL = zero padding.
// ---------------------------------------------------------------------------
static __device__ inline _Float16 haug(const _Float16* __restrict__ S,
                                       int b, int l, int c)
{
    if (l >= LVAL)  return (_Float16)0.f;
    if (c == 512)   return (_Float16)1.f;
    if (c > 512)    return (_Float16)0.f;
    const int i = 1 + l / 63;
    const int j = 1 + (l - (i - 1) * 63);
    size_t idx;
    if (c < 256) idx = ((size_t)(i * WW + (j - 1)) * B_ + b) * NR + c;
    else         idx = ((size_t)((i - 1) * WW + j) * B_ + b) * NR + (c - 256);
    return S[idx];
}

// ---------------------------------------------------------------------------
// HtH[b] = Haug^T @ Haug  (528x528 padded, f32). One wave per row of 4 tiles:
// the A-fragment (tile row t1) is built once per k-block and amortized over
// 4 consecutive t2 tiles. t2 tiles 33..35 hit only zero-padding columns and
// are computed-but-not-stored.
// ---------------------------------------------------------------------------
__global__ __launch_bounds__(32) void esn_hth(const _Float16* __restrict__ S,
                                              float* __restrict__ HtH)
{
    const int t1     = blockIdx.x;
    const int t2base = blockIdx.y * 4;
    const int b      = blockIdx.z;
    const int lane   = threadIdx.x;
    const int laneN  = lane & 15;
    const int half   = lane >> 4;
    const int c1 = t1 * 16 + laneN;   // A row (HtH row dim)

    v8f acc[4];
#pragma unroll
    for (int q = 0; q < 4; ++q) acc[q] = vzero8();

    for (int kb = 0; kb < KB_TOT; ++kb) {
        const int lb = kb * 32;
        FragU af;
#pragma unroll
        for (int v = 0; v < 8; ++v) {
            const int kkA = kmapA(v, half);
            PkU pa;
            pa.h[0] = haug(S, b, lb + kkA,     c1);
            pa.h[1] = haug(S, b, lb + kkA + 1, c1);
            af.u[v] = pa.u;
        }
#pragma unroll
        for (int q = 0; q < 4; ++q) {
            const int c2 = (t2base + q) * 16 + laneN;
            FragU bf;
#pragma unroll
            for (int v = 0; v < 8; ++v) {
                const int kkB = half * 16 + 2 * v;
                PkU pb;
                pb.h[0] = haug(S, b, lb + kkB,     c2);
                pb.h[1] = haug(S, b, lb + kkB + 1, c2);
                bf.u[v] = pb.u;
            }
            acc[q] = __builtin_amdgcn_wmma_f32_16x16x32_f16(
                false, af.v, false, bf.v, (short)0, acc[q], false, false);
        }
    }
#pragma unroll
    for (int q = 0; q < 4; ++q) {
        const int t2 = t2base + q;
        if (t2 >= DPAD / 16) continue;
#pragma unroll
        for (int vr = 0; vr < 8; ++vr) {
            const int r = t1 * 16 + half * 8 + vr;
            HtH[((size_t)b * DPAD + r) * DPAD + t2 * 16 + laneN] = acc[q][vr];
        }
    }
}

// ---------------------------------------------------------------------------
// HtU[b][d] = sum_l Haug[b][l][d] * x[b][1+l/63][1+l%63]
// ---------------------------------------------------------------------------
__global__ __launch_bounds__(544) void esn_htu(const _Float16* __restrict__ S,
                                               const float* __restrict__ x,
                                               float* __restrict__ HtU)
{
    const int b = blockIdx.x;
    const int d = threadIdx.x;
    if (d >= DDIM) return;
    float s = 0.f;
    for (int l = 0; l < LVAL; ++l) {
        const int i = 1 + l / 63;
        const int j = 1 + (l - (i - 1) * 63);
        const float t = x[(size_t)(b * HH + i) * WW + j];
        s += (float)haug(S, b, l, d) * t;
    }
    HtU[(size_t)b * DPAD + d] = s;
}

// ---------------------------------------------------------------------------
// (HtH + 25 I) z = HtU : Gaussian elimination + back substitution, f32,
// one workgroup per batch, pivot row staged in LDS.
// ---------------------------------------------------------------------------
__global__ __launch_bounds__(256) void esn_solve(float* __restrict__ G,
                                                 float* __restrict__ U,
                                                 float* __restrict__ out)
{
    __shared__ float piv[DPAD];
    __shared__ float xs[DPAD];
    __shared__ float red[256];
    __shared__ float sInv, sRk;
    const int b = blockIdx.x;
    const int tid = threadIdx.x;
    float* A   = G + (size_t)b * DPAD * DPAD;
    float* rhs = U + (size_t)b * DPAD;

    for (int d = tid; d < DDIM; d += 256) A[(size_t)d * DPAD + d] += ALPHA2;
    __syncthreads();

    for (int k = 0; k < DDIM; ++k) {
        if (tid == 0) { sInv = 1.0f / A[(size_t)k * DPAD + k]; sRk = rhs[k]; }
        for (int jj = k + tid; jj < DDIM; jj += 256) piv[jj] = A[(size_t)k * DPAD + jj];
        __syncthreads();
        const float inv = sInv, rk = sRk;
        for (int ii = k + 1 + tid; ii < DDIM; ii += 256) {
            const float f = A[(size_t)ii * DPAD + k] * inv;
            for (int jj = k; jj < DDIM; ++jj)
                A[(size_t)ii * DPAD + jj] -= f * piv[jj];
            rhs[ii] -= f * rk;
        }
        __syncthreads();
    }

    for (int k = DDIM - 1; k >= 0; --k) {
        float p = 0.f;
        for (int jj = k + 1 + tid; jj < DDIM; jj += 256)
            p += A[(size_t)k * DPAD + jj] * xs[jj];
        red[tid] = p;
        __syncthreads();
        for (int s = 128; s > 0; s >>= 1) {
            if (tid < s) red[tid] += red[tid + s];
            __syncthreads();
        }
        if (tid == 0) xs[k] = (rhs[k] - red[0]) / A[(size_t)k * DPAD + k];
        __syncthreads();
    }
    for (int d = tid; d < DDIM; d += 256) out[(size_t)b * DDIM + d] = xs[d];
}

// ===========================================================================
extern "C" void kernel_launch(void* const* d_in, const int* in_sizes, int n_in,
                              void* d_out, int out_size, void* d_ws, size_t ws_size,
                              hipStream_t stream)
{
    const float* x   = (const float*)d_in[0];   // (64,64,64)
    const float* Win = (const float*)d_in[1];   // (1,256)
    const float* W1  = (const float*)d_in[2];   // (256,256)
    const float* W2  = (const float*)d_in[3];   // (256,256)
    float* out = (float*)d_out;                 // (64,513)

    char* ws = (char*)d_ws;
    const size_t offS = (size_t)NPIX * B_ * NR * sizeof(_Float16); // 128 MB
    _Float16* S   = (_Float16*)ws;
    unsigned* W1p = (unsigned*)(ws + offS);              // 128 KB
    unsigned* W2p = (unsigned*)(ws + offS + 131072);     // contiguous with W1p
    float*    HtH = (float*)(ws + offS + 262144);
    float*    HtU = (float*)(ws + offS + 262144 + (size_t)B_ * DPAD * DPAD * sizeof(float));

    pack_weights<<<256, 256, 0, stream>>>(W1, W2, W1p, W2p);

    for (int d = 0; d < HH + WW - 1; ++d) {
        const int ilo = (d > WW - 1) ? d - (WW - 1) : 0;
        const int ihi = (d < HH - 1) ? d : HH - 1;
        const int np  = ihi - ilo + 1;
        esn_diag<<<np, 256, 0, stream>>>(x, Win, W1p, W2p, S, d, ilo);
    }

    esn_hth<<<dim3(DPAD / 16, 9, B_), 32, 0, stream>>>(S, HtH);
    esn_htu<<<B_, 544, 0, stream>>>(S, x, HtU);
    esn_solve<<<B_, 256, 0, stream>>>(HtH, HtU, out);
}
